// VectorDiscoConvS2_7404523618488
// MI455X (gfx1250) — matrix operand order repaired and checked by reading, not compiled, not run on hardware
//
#include <hip/hip_runtime.h>

// ================= CDNA5 (gfx1250) WMMA plumbing — wave32 =================
typedef __attribute__((ext_vector_type(16))) __bf16 v16bf;
typedef __attribute__((ext_vector_type(8)))  float  v8f;

#if defined(__AMDGCN__) && __has_builtin(__builtin_amdgcn_global_load_async_to_lds_b128)
#define HAVE_ASYNC_LDS 1
typedef int v4i_ __attribute__((vector_size(16)));
typedef __attribute__((address_space(1))) v4i_ as1_v4i;   // global ("__device__")
typedef __attribute__((address_space(3))) v4i_ as3_v4i;   // LDS ("__shared__")
#else
#define HAVE_ASYNC_LDS 0
#endif

__device__ __forceinline__ void wait_async_lds() {
#if HAVE_ASYNC_LDS
#if __has_builtin(__builtin_amdgcn_s_wait_asynccnt)
  __builtin_amdgcn_s_wait_asynccnt(0);
#else
  asm volatile("s_wait_asynccnt 0" ::: "memory");
#endif
#endif
}

__device__ __forceinline__ unsigned short f2bfu(float f) {
  union { float f; unsigned u; } a; a.f = f;
  unsigned u = a.u;
  return (unsigned short)((u + 0x7FFFu + ((u >> 16) & 1u)) >> 16); // RNE
}

// A-operand (16-bit 16x32 MxK): lane-group hi=(lane>>4); element e -> K offset
__device__ __forceinline__ int akofs(int e, int hi) {
  return hi * 16 + e;
}
// B-operand (16-bit 32x16 KxN): element e -> K offset = hi*16+e; column = lane&15

__device__ __forceinline__ v16bf load_frag(const unsigned short* p) {
  v16bf r;
  __builtin_memcpy(&r, p, 32);   // 32B contiguous -> 2x b128 loads
  return r;
}

// ================= Problem constants =================
#define HH     180
#define WW     360
#define MF     181      // rfft bins
#define NP     368      // padded freq width (23*16), re/im interleaved
#define NT_F   23       // freq N-tiles
#define KB_RF  12       // 384/32 k-steps for rfft/irfft
#define CH     64
#define BWD    7
#define ROWS_F 11520    // 64*180 rows per field
#define ROWS_T 34560    // 3 fields
#define RT_T   2160     // ROWS_T/16 row-tiles
#define ZK     448      // 64 channels * 7 k-slots (14*32)
#define ZKB    14
#define ZSTR   456      // deskewed LDS row stride (ushorts)
#define FRAG   512      // 32 lanes * 16 elements per fragment-tile
#define WG_SZ  28672    // 4 mt * 14 kb * 512 per weight group
#define NSL    11       // psi slices per tile (padded)
#define PSITF  (NSL * BWD * 8 * 2)   // floats per psi tile = 1232 (4928 B)

// ================= Fragment-order table builders =================
__global__ void k_rfft_table(unsigned short* Brf) {
  int idx = blockIdx.x * blockDim.x + threadIdx.x;
  if (idx >= NT_F * KB_RF * FRAG) return;
  int e = idx & 15, t = idx >> 4;
  int l = t & 31; t >>= 5;
  int kb = t % KB_RF, nt = t / KB_RF;
  int kk = kb * 32 + (l >> 4) * 16 + e;
  int n  = nt * 16 + (l & 15);
  float v = 0.f;
  if (kk < WW && n < 2 * MF) {
    int m = n >> 1;
    int p = (m * kk) % WW;
    float ang = 6.283185307179586f * (float)p / (float)WW;
    v = (n & 1) ? -__sinf(ang) : __cosf(ang);
  }
  Brf[idx] = f2bfu(v);
}

__global__ void k_irfft_table(unsigned short* Birf) {
  int idx = blockIdx.x * blockDim.x + threadIdx.x;
  if (idx >= NT_F * KB_RF * FRAG) return;
  int e = idx & 15, t = idx >> 4;
  int l = t & 31; t >>= 5;
  int kb = t % KB_RF, nt = t / KB_RF;
  int kk = kb * 32 + (l >> 4) * 16 + e;
  int w  = nt * 16 + (l & 15);
  float v = 0.f;
  if (kk < 2 * MF && w < WW) {
    int m = kk >> 1;
    float alpha = (m == 0 || m == MF - 1) ? (1.f / WW) : (2.f / WW);
    int p = (m * w) % WW;
    float ang = 6.283185307179586f * (float)p / (float)WW;
    v = (kk & 1) ? (-alpha * __sinf(ang)) : (alpha * __cosf(ang));
  }
  Birf[idx] = f2bfu(v);
}

// Weights in A-fragment order: [grp][mt][kb][lane][e]
__global__ void k_pack_w(const float* w_ss, const float* w_vs,
                         const float* w_sv, const float* w_vv,
                         unsigned short* Wsw) {
  int idx = blockIdx.x * blockDim.x + threadIdx.x;
  if (idx >= 2 * WG_SZ) return;
  int e = idx & 15, t = idx >> 4;
  int l = t & 31; t >>= 5;
  int kb = t % ZKB; t /= ZKB;
  int mt = t % 4, grp = t / 4;
  int o  = mt * 16 + (l & 15);
  int ka = kb * 32 + akofs(e, l >> 4);
  int c = ka / 7, k = ka % 7;
  float v;
  if (grp == 0) v = (k < 3) ? w_ss[(o * CH + c) * 3 + k] : w_vs[(o * CH + c) * 4 + (k - 3)];
  else          v = (k < 4) ? w_sv[(o * CH + c) * 4 + k] : w_vv[(o * CH + c) * 3 + (k - 4)];
  Wsw[idx] = f2bfu(v);
}

// psi tiles: Psit[g][h][mt][slice][w][mi][2]; signs folded; zero-padded
__global__ void k_psipack(const float* __restrict__ psi_s,  const float* __restrict__ psi_cg,
                          const float* __restrict__ psi_sg, const float* __restrict__ psi_cp,
                          const float* __restrict__ psi_sp, const float* __restrict__ psi_cb,
                          const float* __restrict__ psi_sb, float* __restrict__ Psit) {
  int idx = blockIdx.x * blockDim.x + threadIdx.x;
  if (idx >= 3 * HH * NT_F * NSL * BWD * 8) return;
  int mi = idx & 7, t = idx >> 3;
  int w = t % BWD; t /= BWD;
  int slice = t % NSL; t /= NSL;
  int mt = t % NT_F; t /= NT_F;
  int h = t % HH, g = t / HH;
  int m = mt * 8 + mi;
  const float* P = nullptr; int k = 0; float sgn = 1.f;
  if (g == 0) {
    if (slice < 3)      { P = psi_s;  k = slice; }
    else if (slice < 7) { P = psi_cb; k = slice - 3; }
    else                { P = psi_sb; k = slice - 7; }
  } else if (g == 1) {
    if (slice < 4)       { P = psi_cp; k = slice; }
    else if (slice < 7)  { P = psi_cg; k = slice - 4; }
    else if (slice < 10) { P = psi_sg; k = slice - 7; sgn = -1.f; }
  } else {
    if (slice < 4)       { P = psi_sp; k = slice; }
    else if (slice < 7)  { P = psi_sg; k = slice - 4; }
    else if (slice < 10) { P = psi_cg; k = slice - 7; }
  }
  float re = 0.f, im = 0.f;
  if (P && m < MF) {
    const float* p = P + ((size_t)(((k * HH + h) * BWD + w)) * MF + m) * 2;
    re = sgn * p[0]; im = sgn * p[1];
  }
  Psit[(size_t)idx * 2]     = re;
  Psit[(size_t)idx * 2 + 1] = im;
}

// ================= Input convert: f32 -> bf16 in A-fragment order =================
__global__ void k_xconvert(const float* __restrict__ xs, const float* __restrict__ xv,
                           unsigned short* __restrict__ Asw) {
  int idx = blockIdx.x * blockDim.x + threadIdx.x;
  int e = idx & 15, t = idx >> 4;
  int l = t & 31; t >>= 5;
  int kb = t % KB_RF, rt = t / KB_RF;
  int row = rt * 16 + (l & 15);
  int K = kb * 32 + akofs(e, l >> 4);
  int field = row / ROWS_F, rem = row % ROWS_F;
  int c = rem / HH, hrow = rem % HH;
  const float* src = (field == 0)
      ? (xs + (size_t)(c * HH + hrow) * WW)
      : (xv + (size_t)((c * 2 + (field - 1)) * HH + hrow) * WW);
  __builtin_prefetch(src + K + 64, 0, 1);
  Asw[idx] = (K < WW) ? f2bfu(src[K]) : (unsigned short)0;
}

__global__ void k_fconvert(const float* __restrict__ outf, unsigned short* __restrict__ Asw2) {
  int idx = blockIdx.x * blockDim.x + threadIdx.x;
  int e = idx & 15, t = idx >> 4;
  int l = t & 31; t >>= 5;
  int kb = t % KB_RF, rt = t / KB_RF;
  int row = rt * 16 + (l & 15);
  int K = kb * 32 + akofs(e, l >> 4);
  Asw2[idx] = (K < NP) ? f2bfu(outf[(size_t)row * NP + K]) : (unsigned short)0;
}

// ================= Stage 1: rfft as bf16 WMMA GEMM =================
__global__ void __launch_bounds__(128) k_rfft_gemm(const unsigned short* __restrict__ Asw,
                                                   const unsigned short* __restrict__ Brf,
                                                   float* __restrict__ xf) {
  int wv = threadIdx.x >> 5, l = threadIdx.x & 31;
  int lr = l & 15, hi = l >> 4;
  int rt = blockIdx.x * 4 + wv, nt = blockIdx.y;
  const unsigned short* ap = Asw + ((size_t)rt * KB_RF * 32 + l) * 16;
  const unsigned short* bp = Brf + ((size_t)nt * KB_RF * 32 + l) * 16;
  v8f acc = {};
#pragma unroll
  for (int kb = 0; kb < KB_RF; ++kb) {
    v16bf a = load_frag(ap + kb * FRAG);
    v16bf b = load_frag(bp + kb * FRAG);
    acc = __builtin_amdgcn_wmma_f32_16x16x32_bf16(false, a, false, b,
                                                  (short)0, acc, false, false);
  }
  int col = nt * 16 + lr;
#pragma unroll
  for (int j = 0; j < 8; ++j)
    xf[(size_t)(rt * 16 + hi * 8 + j) * NP + col] = acc[j];
}

// ================= Stage 2: banded complex taps (VALU) + WMMA mix =================
// psi tile (4928 B) copy: async global->LDS when available, overlapped double-buffer
__device__ __forceinline__ void copy_psi_tile(const float* __restrict__ g, float* l, int tid) {
  for (int t = tid; t < PSITF / 4; t += 128) {       // 308 x 16B
#if HAVE_ASYNC_LDS
    __builtin_amdgcn_global_load_async_to_lds_b128(
        (as1_v4i*)(g + t * 4), (as3_v4i*)(l + t * 4), 0, 0);
#else
    float4 tmp = *(const float4*)(g + t * 4);
    *(float4*)(l + t * 4) = tmp;
#endif
  }
}

__device__ __forceinline__ void disco_acc_l(const float* __restrict__ ps,   // slice base in LDS
                                            const float* __restrict__ xf_c,
                                            const int* gh, int mi, int m,
                                            float& yr, float& yi) {
#pragma unroll
  for (int w = 0; w < BWD; ++w) {
    float pr = ps[w * 16 + 2 * mi];
    float pi = ps[w * 16 + 2 * mi + 1];
    const float* x = xf_c + (size_t)gh[w] * NP + 2 * m;
    float xr = x[0], xi = x[1];
    yr += pr * xr - pi * xi;
    yi += pr * xi + pi * xr;
  }
}

__global__ void __launch_bounds__(128) k_disco(const float* __restrict__ xf,
    const float* __restrict__ Psit, const int* __restrict__ gather,
    const unsigned short* __restrict__ Wsw, float* __restrict__ outf) {
  __shared__ unsigned short Zs[16 * ZSTR];   // transposed Z, deskewed
  __shared__ float psiS[2][PSITF];           // double-buffered psi tile
  __shared__ int gh[BWD];
  int h = blockIdx.x, g = blockIdx.y;
  int tid = threadIdx.x;
  if (tid < BWD) gh[tid] = gather[h * BWD + tid];
  int wv = tid >> 5, l = tid & 31, lr = l & 15, hi = l >> 4;
  int gsel = (g == 0) ? 0 : 1;
  const float* ptile = Psit + (size_t)((g * HH + h) * NT_F) * PSITF;

  // Hoist all 14 weight A-fragments (read once per wave).
  v16bf afr[ZKB];
  {
    const unsigned short* wp = Wsw + (size_t)gsel * WG_SZ + ((size_t)wv * ZKB * 32 + l) * 16;
#pragma unroll
    for (int kb = 0; kb < ZKB; ++kb) afr[kb] = load_frag(wp + kb * FRAG);
  }

  copy_psi_tile(ptile, psiS[0], tid);        // prefetch tile 0

  for (int mt = 0; mt < NT_F; ++mt) {
    wait_async_lds();
    __syncthreads();                         // psi tile ready; Zs free for rewrite
    const float* pc = psiS[mt & 1];
    if (mt + 1 < NT_F)
      copy_psi_tile(ptile + (size_t)(mt + 1) * PSITF, psiS[(mt + 1) & 1], tid);

    // ---- Phase A: Z[c*7+k][2mi(+1)] complex taps (VALU f32, psi from LDS) ----
    for (int idx = tid; idx < ZK * 8; idx += 128) {
      int rowz = idx % ZK, mi = idx / ZK;
      int c = rowz / 7, k = rowz % 7;
      int m = mt * 8 + mi;                   // psi/xf are zero-padded past m=180
      float yr = 0.f, yi = 0.f;
      const float* xs_c = xf + (size_t)(c * HH) * NP;
      const float* xu_c = xf + (size_t)(ROWS_F + c * HH) * NP;
      const float* xv_c = xf + (size_t)(2 * ROWS_F + c * HH) * NP;
      if (g == 0) {
        if (k < 3) disco_acc_l(pc + k * 112, xs_c, gh, mi, m, yr, yi);
        else { disco_acc_l(pc + k * 112,       xu_c, gh, mi, m, yr, yi);
               disco_acc_l(pc + (k + 4) * 112, xv_c, gh, mi, m, yr, yi); }
      } else {
        if (k < 4) disco_acc_l(pc + k * 112, xs_c, gh, mi, m, yr, yi);
        else { disco_acc_l(pc + k * 112,       xu_c, gh, mi, m, yr, yi);
               disco_acc_l(pc + (k + 3) * 112, xv_c, gh, mi, m, yr, yi); }
      }
      Zs[(2 * mi) * ZSTR + rowz]     = f2bfu(yr);
      Zs[(2 * mi + 1) * ZSTR + rowz] = f2bfu(yi);
    }
    __syncthreads();

    // ---- Phase B: out_f tile = W(16x448) * Z(448x16), 14 WMMA k-steps ----
    v8f acc = {};
#pragma unroll
    for (int kb = 0; kb < ZKB; ++kb) {
      v16bf b = load_frag(&Zs[lr * ZSTR + kb * 32 + hi * 16]);
      acc = __builtin_amdgcn_wmma_f32_16x16x32_bf16(false, afr[kb], false, b,
                                                    (short)0, acc, false, false);
    }
#pragma unroll
    for (int j = 0; j < 8; ++j) {
      int o = wv * 16 + hi * 8 + j;
      outf[(size_t)((g * CH + o) * HH + h) * NP + mt * 16 + lr] = acc[j];
    }
  }
}

// ================= Stage 3: irfft as bf16 WMMA GEMM + bias + scatter =================
__global__ void __launch_bounds__(128) k_irfft(const unsigned short* __restrict__ Asw2,
                                               const unsigned short* __restrict__ Birf,
                                               const float* __restrict__ b_s,
                                               float* __restrict__ out) {
  int wv = threadIdx.x >> 5, l = threadIdx.x & 31;
  int lr = l & 15, hi = l >> 4;
  int rt = blockIdx.x * 4 + wv, nt = blockIdx.y;
  const unsigned short* ap = Asw2 + ((size_t)rt * KB_RF * 32 + l) * 16;
  const unsigned short* bp = Birf + ((size_t)nt * KB_RF * 32 + l) * 16;
  v8f acc = {};
#pragma unroll
  for (int kb = 0; kb < KB_RF; ++kb) {
    v16bf a = load_frag(ap + kb * FRAG);
    v16bf b = load_frag(bp + kb * FRAG);
    acc = __builtin_amdgcn_wmma_f32_16x16x32_bf16(false, a, false, b,
                                                  (short)0, acc, false, false);
  }
  int col = nt * 16 + lr;
  if (col < WW) {
#pragma unroll
    for (int j = 0; j < 8; ++j) {
      int r2 = rt * 16 + hi * 8 + j;
      int grp = r2 / ROWS_F, rem = r2 % ROWS_F;
      int o = rem / HH, hrow = rem % HH;
      float v = acc[j];
      int ch;
      if (grp == 0) { v += b_s[o]; ch = o; }
      else          { ch = CH + 2 * o + (grp - 1); }
      out[(size_t)(ch * HH + hrow) * WW + col] = v;
    }
  }
}

// ================= Host launch =================
extern "C" void kernel_launch(void* const* d_in, const int* in_sizes, int n_in,
                              void* d_out, int out_size, void* d_ws, size_t ws_size,
                              hipStream_t stream) {
  (void)in_sizes; (void)n_in; (void)out_size; (void)ws_size;
  const float* xs     = (const float*)d_in[0];
  const float* xv     = (const float*)d_in[1];
  const float* w_ss   = (const float*)d_in[2];
  const float* w_vs   = (const float*)d_in[3];
  const float* w_sv   = (const float*)d_in[4];
  const float* w_vv   = (const float*)d_in[5];
  const float* b_s    = (const float*)d_in[6];
  const float* psi_s  = (const float*)d_in[7];
  const float* psi_cg = (const float*)d_in[8];
  const float* psi_sg = (const float*)d_in[9];
  const float* psi_cp = (const float*)d_in[10];
  const float* psi_sp = (const float*)d_in[11];
  const float* psi_cb = (const float*)d_in[12];
  const float* psi_sb = (const float*)d_in[13];
  const int*   gather = (const int*)d_in[14];

  char* ws = (char*)d_ws;
  size_t off = 0;
  auto alloc = [&](size_t bytes) -> char* {
    char* p = ws + off;
    off = (off + bytes + 255) & ~(size_t)255;
    return p;
  };
  const size_t tabE  = (size_t)NT_F * KB_RF * FRAG;
  const size_t aswE  = (size_t)RT_T * KB_RF * FRAG;
  const size_t psitE = (size_t)3 * HH * NT_F * PSITF;      // floats
  unsigned short* Brf  = (unsigned short*)alloc(tabE * 2);
  unsigned short* Birf = (unsigned short*)alloc(tabE * 2);
  unsigned short* Wsw  = (unsigned short*)alloc((size_t)2 * WG_SZ * 2);
  unsigned short* Asw  = (unsigned short*)alloc(aswE * 2);
  unsigned short* Asw2 = (unsigned short*)alloc(aswE * 2);
  float* Psit = (float*)alloc(psitE * 4);
  float* xf   = (float*)alloc((size_t)ROWS_T * NP * 4);
  float* outf = (float*)alloc((size_t)ROWS_T * NP * 4);

  k_rfft_table <<<(int)((tabE + 255) / 256), 256, 0, stream>>>(Brf);
  k_irfft_table<<<(int)((tabE + 255) / 256), 256, 0, stream>>>(Birf);
  k_pack_w<<<(2 * WG_SZ + 255) / 256, 256, 0, stream>>>(w_ss, w_vs, w_sv, w_vv, Wsw);
  k_psipack<<<(int)((psitE / 2 + 255) / 256), 256, 0, stream>>>(psi_s, psi_cg, psi_sg,
                                                                psi_cp, psi_sp, psi_cb,
                                                                psi_sb, Psit);
  k_xconvert<<<(int)(aswE / 256), 256, 0, stream>>>(xs, xv, Asw);

  k_rfft_gemm<<<dim3(RT_T / 4, NT_F), 128, 0, stream>>>(Asw, Brf, xf);
  k_disco<<<dim3(HH, 3), 128, 0, stream>>>(xf, Psit, gather, Wsw, outf);
  k_fconvert<<<(int)(aswE / 256), 256, 0, stream>>>(outf, Asw2);
  k_irfft<<<dim3(RT_T / 4, NT_F), 128, 0, stream>>>(Asw2, Birf, b_s, (float*)d_out);
}
